// TemporalRefinementNetwork_28132035789044
// MI455X (gfx1250) — matrix operand order repaired and checked by reading, not compiled, not run on hardware
//
#include <hip/hip_runtime.h>
#include <hip/hip_bf16.h>
#include <math.h>

// ---------------------------------------------------------------------------
// TemporalRefinementNetwork forward for MI455X (gfx1250, wave32, WMMA).
// Shapes (fixed by setup_inputs): B=1 Q=128 T=48 C=64 H=96 W=128 D=128 NB=4.
// GEMMs run in f16 (f32 accum) on v_wmma_f32_16x16x32_f16.
// Correlation stage uses global_load_async_to_lds_b32 (ASYNCcnt) to stage an
// 8x8x64ch patch in LDS, then factorizes the 49-tap bilinear correlation
// through a per-pixel channel reduction.
// ---------------------------------------------------------------------------

typedef __attribute__((ext_vector_type(16))) _Float16 v16h;
typedef __attribute__((ext_vector_type(8)))  _Float16 v8h;
typedef __attribute__((ext_vector_type(8)))  float    v8f;

#define BQ    128
#define T_N   48
#define C_N   64
#define H_N   96
#define W_N   128
#define D_N   128
#define NBLK  4
#define ROWS  (BQ * T_N)          /* 6144 */
#define KPAD  288                 /* 276 padded to multiple of 32 */
#define EPSV  1e-5f

__device__ __forceinline__ float gelu_exact(float v) {
  return 0.5f * v * (1.0f + erff(v * 0.70710678118654752f));
}

// ------------------------- init / finalize ---------------------------------

__global__ void init_state(const float* __restrict__ coarse,
                           float* __restrict__ tracks,
                           float* __restrict__ x,
                           float* __restrict__ vis_sum) {
  int i = blockIdx.x * blockDim.x + threadIdx.x;
  if (i < ROWS * 2)   tracks[i]  = coarse[i];
  if (i < ROWS)       vis_sum[i] = 0.0f;
  if (i < ROWS * D_N) x[i]       = 0.0f;
}

__global__ void finalize_out(const float* __restrict__ tracks,
                             const float* __restrict__ vis_sum,
                             float* __restrict__ out) {
  int i = blockIdx.x * blockDim.x + threadIdx.x;
  if (i >= ROWS) return;
  out[i * 3 + 0] = tracks[i * 2 + 0];
  out[i * 3 + 1] = tracks[i * 2 + 1];
  out[i * 3 + 2] = vis_sum[i] * 0.25f;   // / NUM_ITERS
}

// ------------------ weight convert + transpose to f16 ----------------------

// w[276][128] (row-major) -> wT[128][KPAD] padded with zeros
__global__ void convert_projT(const float* __restrict__ w, _Float16* __restrict__ wT) {
  int i = blockIdx.x * blockDim.x + threadIdx.x;
  if (i >= D_N * KPAD) return;
  int n = i / KPAD, k = i % KPAD;
  wT[i] = (k < 276) ? (_Float16)w[k * D_N + n] : (_Float16)0.0f;
}

// w[Kd][N] -> wT[N][Kd]
__global__ void convert_T(const float* __restrict__ w, _Float16* __restrict__ wT,
                          int Kd, int N) {
  int i = blockIdx.x * blockDim.x + threadIdx.x;
  if (i >= Kd * N) return;
  int n = i / Kd, k = i % Kd;
  wT[i] = (_Float16)w[k * N + n];
}

// --------------- stage 1: corr + fourier + hidden -> Xin(f16) --------------
//
// All 49 taps share one fractional offset, so every bilinear corner lies in
// an 8x8 pixel window [X0..X0+7]x[Y0..Y0+7].  We async-copy the 64ch x 8x8
// patch into LDS, reduce over channels once (pdot[64]), then each tap is a
// 4-point combine.  Out-of-range pixels are clamped on load but masked to
// zero weight in the combine (matches the reference's valid-mask semantics).

__global__ void stage1(const float* __restrict__ fmaps,
                       const float* __restrict__ qfeat,
                       const float* __restrict__ tracks,
                       const float* __restrict__ hidden,
                       _Float16* __restrict__ Xin) {
  int bqt = blockIdx.x;                 // q*T + t
  int q = bqt / T_N, t = bqt % T_N;
  int tid = threadIdx.x;                // 0..127

  __shared__ float patch[C_N * 64];     // [c][py*8+px], 16 KB
  __shared__ float pdot[64];
  __shared__ float qf[C_N];
  __shared__ float trkS[2];
  if (tid < C_N) qf[tid] = qfeat[q * C_N + tid];
  if (tid < 2)   trkS[tid] = tracks[bqt * 2 + tid];
  __syncthreads();

  float cx = trkS[0] * 0.25f;           // feature_stride = 4
  float cy = trkS[1] * 0.25f;
  float xf = floorf(cx), yf = floorf(cy);
  float fx = cx - xf,    fy = cy - yf;
  int X0 = (int)xf - 3, Y0 = (int)yf - 3;

  const float* f = fmaps + (size_t)t * C_N * H_N * W_N;

  // ---- async patch fill: 4096 elements, 32 per thread (ASYNCcnt path) ----
  #pragma unroll 4
  for (int it = 0; it < 32; ++it) {
    int e  = tid + it * 128;
    int c  = e >> 6;
    int px = e & 7;
    int py = (e >> 3) & 7;
    int gx = min(max(X0 + px, 0), W_N - 1);
    int gy = min(max(Y0 + py, 0), H_N - 1);
    const float* gp = f + (size_t)c * H_N * W_N + (size_t)gy * W_N + gx;
    unsigned int       lds = (unsigned int)(uintptr_t)(&patch[e]);   // low 32b = LDS offset
    unsigned long long ga  = (unsigned long long)(uintptr_t)gp;
    asm volatile("global_load_async_to_lds_b32 %0, %1, off"
                 :: "v"(lds), "v"(ga) : "memory");
  }
  asm volatile("s_wait_asynccnt 0x0" ::: "memory");
  __syncthreads();

  // ---- per-pixel channel reduction: pdot[py*8+px] = <patch[:,p], qf> ----
  if (tid < 64) {
    float s = 0.0f;
    #pragma unroll 8
    for (int c = 0; c < C_N; ++c) s += patch[c * 64 + tid] * qf[c];
    pdot[tid] = s;
  }
  __syncthreads();

  _Float16* xrow = Xin + (size_t)bqt * KPAD;

  // ---- 49 taps: bilinear combine of pdot with valid masks ----
  if (tid < 49) {
    int lx = tid % 7;                   // local corner base (offset -3 maps to 0)
    int ly = tid / 7;
    int gx0 = X0 + lx, gy0 = Y0 + ly;
    float v00 = (gx0     >= 0 && gx0     < W_N && gy0     >= 0 && gy0     < H_N) ? 1.f : 0.f;
    float v10 = (gx0 + 1 >= 0 && gx0 + 1 < W_N && gy0     >= 0 && gy0     < H_N) ? 1.f : 0.f;
    float v01 = (gx0     >= 0 && gx0     < W_N && gy0 + 1 >= 0 && gy0 + 1 < H_N) ? 1.f : 0.f;
    float v11 = (gx0 + 1 >= 0 && gx0 + 1 < W_N && gy0 + 1 >= 0 && gy0 + 1 < H_N) ? 1.f : 0.f;
    float dot = (1.f - fx) * (1.f - fy) * v00 * pdot[ly * 8 + lx]
              +        fx  * (1.f - fy) * v10 * pdot[ly * 8 + lx + 1]
              + (1.f - fx) *        fy  * v01 * pdot[(ly + 1) * 8 + lx]
              +        fx  *        fy  * v11 * pdot[(ly + 1) * 8 + lx + 1];
    xrow[tid] = (_Float16)dot;
  }

  // ---- fourier positional encoding (threads 0..98) ----
  if (tid < 99) {
    float coords3[3] = { trkS[0], trkS[1], (float)t / (float)(T_N - 1) };
    float val;
    if (tid < 3) {
      val = coords3[tid];
    } else {
      int rel = tid - 3;
      int ci = rel >> 5;                    // which coord
      int w32 = rel & 31;
      int band = w32 & 15;
      bool isCos = (w32 >= 16);
      float freq = 3.14159265358979f * exp2f((float)band * (9.0f / 15.0f));
      float a = coords3[ci] * freq;
      val = isCos ? cosf(a) : sinf(a);
    }
    xrow[49 + tid] = (_Float16)val;
  }

  // ---- hidden (128 values) + pad ----
  xrow[148 + tid] = (_Float16)hidden[(size_t)bqt * D_N + tid];
  if (tid < KPAD - 276) xrow[276 + tid] = (_Float16)0.0f;
}

// ------------------------------ WMMA GEMM ----------------------------------
// C[M,N] = A[M,Kd](f16) * BT[N,Kd]^T(f16) (+bias) (+GELU) (+residual)
// Each wave computes a 16x64 strip of C (A fragment reused over 4 N-tiles).
// flags: 1 = GELU, 2 = residual add into Cf, 4 = write f32 Cf, 8 = write f16 Ch

__device__ __forceinline__ v16h load_frag_a(const _Float16* __restrict__ A,
                                            int lda, int m0, int k0, int lane) {
  int row = m0 + (lane & 15);
  int kb  = k0 + ((lane >> 4) << 3);          // lanes 16-31 take K+8 / K+24
  const _Float16* p = A + (size_t)row * lda + kb;
  v8h lo = *(const v8h*)p;                    // K = kb .. kb+7
  v8h hi = *(const v8h*)(p + 16);             // K = kb+16 .. kb+23
  return __builtin_shufflevector(lo, hi, 0,1,2,3,4,5,6,7,8,9,10,11,12,13,14,15);
}

__device__ __forceinline__ v16h load_frag_b(const _Float16* __restrict__ BT,
                                            int ldb, int n0, int k0, int lane) {
  int col = n0 + (lane & 15);
  int kb  = k0 + ((lane >> 4) << 3);          // lanes 16-31 take K+8 / K+24
  const _Float16* p = BT + (size_t)col * ldb + kb;
  v8h lo = *(const v8h*)p;
  v8h hi = *(const v8h*)(p + 16);
  return __builtin_shufflevector(lo, hi, 0,1,2,3,4,5,6,7,8,9,10,11,12,13,14,15);
}

__global__ void gemm_wmma_f16(const _Float16* __restrict__ A,
                              const _Float16* __restrict__ BT,
                              const float* __restrict__ bias,
                              float* __restrict__ Cf,
                              _Float16* __restrict__ Ch,
                              int M, int N, int Kd, int flags) {
  int lane = threadIdx.x & 31;
  int wave = threadIdx.x >> 5;
  int Ntw  = N >> 6;                          // 64-wide strips
  int Mt   = M >> 4;
  int tile = blockIdx.x * (blockDim.x >> 5) + wave;
  if (tile >= Mt * Ntw) return;               // wave-uniform: EXEC stays all-1s
  int tm = tile / Ntw;
  int n0 = (tile % Ntw) << 6;
  int m0 = tm << 4;

  v8f acc0 = {}, acc1 = {}, acc2 = {}, acc3 = {};
  for (int k0 = 0; k0 < Kd; k0 += 32) {
    if (k0 + 32 < Kd) {                       // WGP-scope global_prefetch_b8
      __builtin_prefetch(A + (size_t)(m0 + (lane & 15)) * Kd + k0 + 32, 0, 3);
      __builtin_prefetch(BT + (size_t)(n0 + (lane & 15)) * Kd + k0 + 32, 0, 3);
    }
    v16h a  = load_frag_a(A, Kd, m0, k0, lane);
    v16h b0 = load_frag_b(BT, Kd, n0 +  0, k0, lane);
    v16h b1 = load_frag_b(BT, Kd, n0 + 16, k0, lane);
    v16h b2 = load_frag_b(BT, Kd, n0 + 32, k0, lane);
    v16h b3 = load_frag_b(BT, Kd, n0 + 48, k0, lane);
    acc0 = __builtin_amdgcn_wmma_f32_16x16x32_f16(false, a, false, b0, (short)0, acc0, false, false);
    acc1 = __builtin_amdgcn_wmma_f32_16x16x32_f16(false, a, false, b1, (short)0, acc1, false, false);
    acc2 = __builtin_amdgcn_wmma_f32_16x16x32_f16(false, a, false, b2, (short)0, acc2, false, false);
    acc3 = __builtin_amdgcn_wmma_f32_16x16x32_f16(false, a, false, b3, (short)0, acc3, false, false);
  }

  int rbase = m0 + ((lane >> 4) << 3);        // lanes 0-15: M=m0..+7, 16-31: +8..+15
  int cbase = lane & 15;
  v8f accs[4] = { acc0, acc1, acc2, acc3 };
  #pragma unroll
  for (int s = 0; s < 4; ++s) {
    int col = n0 + s * 16 + cbase;
    float bv = bias ? bias[col] : 0.0f;
    #pragma unroll
    for (int r = 0; r < 8; ++r) {
      int row = rbase + r;
      size_t idx = (size_t)row * N + col;
      float v = accs[s][r] + bv;
      if (flags & 1) v = gelu_exact(v);
      if (flags & 2) v += Cf[idx];
      if (flags & 8) Ch[idx] = (_Float16)v;
      if (flags & 4) Cf[idx] = v;
    }
  }
}

// -------------------------- layernorm helpers ------------------------------

__device__ __forceinline__ float blockReduceSum128(float v, float* red) {
  #pragma unroll
  for (int off = 16; off > 0; off >>= 1) v += __shfl_xor(v, off, 32);
  int tid = threadIdx.x;
  if ((tid & 31) == 0) red[tid >> 5] = v;
  __syncthreads();
  float total = red[0] + red[1] + red[2] + red[3];
  __syncthreads();
  return total;
}

// Fused LN1 + depthwise temporal conv (ksz=5, pad=2) + residual, per (b,q).
__global__ void ln_conv(float* __restrict__ x,
                        const float* __restrict__ g, const float* __restrict__ b,
                        const float* __restrict__ cw, const float* __restrict__ cb) {
  __shared__ float hln[T_N][D_N];             // 24 KB
  __shared__ float red[4];
  int bq = blockIdx.x, d = threadIdx.x;
  float gv = g[d], bv = b[d];
  for (int t = 0; t < T_N; ++t) {
    float v = x[((size_t)bq * T_N + t) * D_N + d];
    float m = blockReduceSum128(v, red) * (1.0f / D_N);
    float dv = v - m;
    float var = blockReduceSum128(dv * dv, red) * (1.0f / D_N);
    hln[t][d] = dv * rsqrtf(var + EPSV) * gv + bv;
  }
  __syncthreads();
  float w[5];
  #pragma unroll
  for (int j = 0; j < 5; ++j) w[j] = cw[d * 5 + j];
  float cbv = cb[d];
  for (int t = 0; t < T_N; ++t) {
    float acc = cbv;
    #pragma unroll
    for (int j = 0; j < 5; ++j) {
      int tt = t + j - 2;
      if (tt >= 0 && tt < T_N) acc += hln[tt][d] * w[j];
    }
    x[((size_t)bq * T_N + t) * D_N + d] += acc;
  }
}

// Row layernorm -> f16 output (GEMM A operand)
__global__ void ln_rows(const float* __restrict__ x,
                        const float* __restrict__ g, const float* __restrict__ b,
                        _Float16* __restrict__ out) {
  __shared__ float red[4];
  int row = blockIdx.x, d = threadIdx.x;
  float v = x[(size_t)row * D_N + d];
  float m = blockReduceSum128(v, red) * (1.0f / D_N);
  float dv = v - m;
  float var = blockReduceSum128(dv * dv, red) * (1.0f / D_N);
  out[(size_t)row * D_N + d] = (_Float16)(dv * rsqrtf(var + EPSV) * g[d] + b[d]);
}

// ------------------------------ heads --------------------------------------

__global__ void heads(const float* __restrict__ x,
                      const float* __restrict__ dw, const float* __restrict__ db,
                      const float* __restrict__ vw1, const float* __restrict__ vb1,
                      const float* __restrict__ vw2, const float* __restrict__ vb2,
                      float* __restrict__ tracks, float* __restrict__ vis_sum) {
  int row = blockIdx.x * blockDim.x + threadIdx.x;
  if (row >= ROWS) return;
  const float* xr = x + (size_t)row * D_N;
  float d0 = db[0], d1 = db[1];
  float hv[64];
  #pragma unroll
  for (int j = 0; j < 64; ++j) hv[j] = vb1[j];
  for (int c = 0; c < D_N; ++c) {
    float xv = xr[c];
    d0 += xv * dw[c * 2 + 0];
    d1 += xv * dw[c * 2 + 1];
    const float* wrow = vw1 + (size_t)c * 64;
    #pragma unroll
    for (int j = 0; j < 64; ++j) hv[j] += xv * wrow[j];
  }
  float vis = vb2[0];
  #pragma unroll
  for (int j = 0; j < 64; ++j) vis += gelu_exact(hv[j]) * vw2[j];
  tracks[row * 2 + 0] += d0;
  tracks[row * 2 + 1] += d1;
  vis_sum[row] += vis;
}

// ------------------------------ host side ----------------------------------

extern "C" void kernel_launch(void* const* d_in, const int* in_sizes, int n_in,
                              void* d_out, int out_size, void* d_ws, size_t ws_size,
                              hipStream_t stream) {
  const float* coarse = (const float*)d_in[0];
  const float* qfeat  = (const float*)d_in[1];
  const float* fmaps  = (const float*)d_in[2];
  /* d_in[3] = feature_stride (int, ==4) — folded into stage1 as *0.25f */
  const float* proj_w = (const float*)d_in[4];
  const float* proj_b = (const float*)d_in[5];
  const float* ln1_g  = (const float*)d_in[6];
  const float* ln1_b  = (const float*)d_in[7];
  const float* conv_w = (const float*)d_in[8];
  const float* conv_b = (const float*)d_in[9];
  const float* ln2_g  = (const float*)d_in[10];
  const float* ln2_b  = (const float*)d_in[11];
  const float* mlp_w1 = (const float*)d_in[12];
  const float* mlp_b1 = (const float*)d_in[13];
  const float* mlp_w2 = (const float*)d_in[14];
  const float* mlp_b2 = (const float*)d_in[15];
  const float* delta_w = (const float*)d_in[16];
  const float* delta_b = (const float*)d_in[17];
  const float* vis_w1  = (const float*)d_in[18];
  const float* vis_b1  = (const float*)d_in[19];
  const float* vis_w2  = (const float*)d_in[20];
  const float* vis_b2  = (const float*)d_in[21];
  float* out = (float*)d_out;

  // -------- workspace layout (all segments 256B-aligned) --------
  char* ws = (char*)d_ws;
  size_t off = 0;
  auto take = [&](size_t bytes) { char* p = ws + off; off += (bytes + 255) & ~(size_t)255; return p; };
  float*    xf32    = (float*)   take((size_t)ROWS * D_N * 4);   // activation / hidden
  float*    tracks  = (float*)   take((size_t)ROWS * 2 * 4);
  float*    vis_sum = (float*)   take((size_t)ROWS * 4);
  _Float16* Xin     = (_Float16*)take((size_t)ROWS * KPAD * 2);
  _Float16* ln2out  = (_Float16*)take((size_t)ROWS * D_N * 2);
  _Float16* h1      = (_Float16*)take((size_t)ROWS * 2 * D_N * 2);
  _Float16* projT   = (_Float16*)take((size_t)D_N * KPAD * 2);
  _Float16* w1T     = (_Float16*)take((size_t)NBLK * 2 * D_N * D_N * 2);
  _Float16* w2T     = (_Float16*)take((size_t)NBLK * 2 * D_N * D_N * 2);
  (void)ws_size; (void)in_sizes; (void)n_in; (void)out_size;

  // -------- one-time (per launch) weight conversion --------
  {
    int n = D_N * KPAD;
    convert_projT<<<(n + 255) / 256, 256, 0, stream>>>(proj_w, projT);
    for (int i = 0; i < NBLK; ++i) {
      int n1 = D_N * 2 * D_N;  // 128*256
      convert_T<<<(n1 + 255) / 256, 256, 0, stream>>>(
          mlp_w1 + (size_t)i * D_N * 2 * D_N, w1T + (size_t)i * 2 * D_N * D_N, D_N, 2 * D_N);
      convert_T<<<(n1 + 255) / 256, 256, 0, stream>>>(
          mlp_w2 + (size_t)i * 2 * D_N * D_N, w2T + (size_t)i * 2 * D_N * D_N, 2 * D_N, D_N);
    }
  }

  // -------- state init --------
  {
    int n = ROWS * D_N;
    init_state<<<(n + 255) / 256, 256, 0, stream>>>(coarse, tracks, xf32, vis_sum);
  }

  // -------- 4 refinement iterations --------
  for (int it = 0; it < 4; ++it) {
    // corr + fourier + hidden -> Xin (f16, KPAD)
    stage1<<<ROWS, 128, 0, stream>>>(fmaps, qfeat, tracks, xf32, Xin);

    // x = Xin @ proj_w + proj_b          (M=6144, N=128, K=288)
    {
      int waves = (ROWS / 16) * (D_N / 64);   // 768
      gemm_wmma_f16<<<waves / 8, 256, 0, stream>>>(Xin, projT, proj_b,
                                                   xf32, nullptr,
                                                   ROWS, D_N, KPAD, /*write f32*/4);
    }

    for (int i = 0; i < NBLK; ++i) {
      // LN1 + depthwise conv + residual (in place on x)
      ln_conv<<<BQ, D_N, 0, stream>>>(xf32, ln1_g + i * D_N, ln1_b + i * D_N,
                                      conv_w + (size_t)i * D_N * 5, conv_b + i * D_N);
      // LN2 -> f16
      ln_rows<<<ROWS, D_N, 0, stream>>>(xf32, ln2_g + i * D_N, ln2_b + i * D_N, ln2out);
      // h1 = gelu(ln2out @ w1 + b1)      (M=6144, N=256, K=128) -> f16
      {
        int waves = (ROWS / 16) * (2 * D_N / 64);   // 1536
        gemm_wmma_f16<<<waves / 8, 256, 0, stream>>>(ln2out, w1T + (size_t)i * 2 * D_N * D_N,
                                                     mlp_b1 + i * 2 * D_N,
                                                     nullptr, h1,
                                                     ROWS, 2 * D_N, D_N, /*gelu|f16*/1 | 8);
      }
      // x += h1 @ w2 + b2                (M=6144, N=128, K=256)
      {
        int waves = (ROWS / 16) * (D_N / 64);       // 768
        gemm_wmma_f16<<<waves / 8, 256, 0, stream>>>(h1, w2T + (size_t)i * 2 * D_N * D_N,
                                                     mlp_b2 + i * D_N,
                                                     xf32, nullptr,
                                                     ROWS, D_N, 2 * D_N, /*resid|f32*/2 | 4);
      }
    }

    // delta / vis heads; tracks += delta; vis_sum += vis
    heads<<<(ROWS + 63) / 64, 64, 0, stream>>>(xf32, delta_w, delta_b,
                                               vis_w1, vis_b1, vis_w2, vis_b2,
                                               tracks, vis_sum);
  }

  finalize_out<<<(ROWS + 255) / 256, 256, 0, stream>>>(tracks, vis_sum, out);
}
